// LSTMModel_32332513804491
// MI455X (gfx1250) — compile-verified, matrix operand order checked by reading, not run on hardware
//
#include <hip/hip_runtime.h>
#include <hip/hip_bf16.h>

typedef __attribute__((ext_vector_type(16))) _Float16 v16h;
typedef __attribute__((ext_vector_type(8)))  float    v8f;

#define TSTEPS   128
#define HID      128
#define GATES    512
#define BTILE    16
#define BATCH    2048
#define NTHREADS 512
#define NWAVES   (NTHREADS/32)       // 16 waves
#define NTPW     (GATES/16/NWAVES)   // 2 N-tiles (of 16 gate cols) per wave
#define SGPITCH  20                  // floats; pitch 20 -> conflict-free b128

#define LOG2E 1.44269504088896340736f

// ---- fast activations on the gfx1250 TRANS pipe ----
__device__ __forceinline__ float fast_exp2(float x) {
  return __builtin_amdgcn_exp2f(x);          // v_exp_f32
}
__device__ __forceinline__ float fast_sigmoid(float x) {
  // 1/(1+e^-x) = rcp(1 + 2^(-x*log2e)) : v_exp_f32 + v_rcp_f32
  return __builtin_amdgcn_rcpf(1.0f + fast_exp2(-x * LOG2E));
}
__device__ __forceinline__ float fast_tanh(float x) {
#if __has_builtin(__builtin_amdgcn_tanhf)
  return __builtin_amdgcn_tanhf(x);          // v_tanh_f32 (gfx1250)
#else
  float e2 = fast_exp2(2.0f * LOG2E * x);    // e^(2x)
  return 1.0f - 2.0f * __builtin_amdgcn_rcpf(e2 + 1.0f);
#endif
}

__device__ __forceinline__ v8f wmma_f16(v16h a, v16h b, v8f c) {
  // v_wmma_f32_16x16x32_f16: D = A(16x32 f16) * B(32x16 f16) + C(16x16 f32)
  return __builtin_amdgcn_wmma_f32_16x16x32_f16(false, a, false, b, (short)0, c,
                                                false, false);
}

// A-matrix fragment (16x32 f16, ISA 7.12.2 layout): lane m = lane%16 holds row M,
// halves 0..7  cover K = kbase + (laneHalf?8:0) + 0..7,
// halves 8..15 cover K = kbase + 16 + (laneHalf?8:0) + 0..7.
// Each lane fetches its 16 halfs as two 16-byte LDS loads (ds_load_b128 x2).
__device__ __forceinline__ v16h load_a_frag(const _Float16* base, int ld, int m,
                                            int kbase, int laneHalf) {
  const _Float16* p = base + (size_t)m * ld + kbase + (laneHalf ? 8 : 0);
  union { v16h v; uint4 q[2]; } u;
  u.q[0] = *(const uint4*)(p);
  u.q[1] = *(const uint4*)(p + 16);
  return u.v;
}

template <int IN_DIM, int IN_PAD, int FIRST, int LAST>
__global__ __launch_bounds__(NTHREADS) void lstm_layer_kernel(
    const float*    __restrict__ x_f32,   // FIRST: x  (B, T, IN_DIM) f32
    const _Float16* __restrict__ x_f16,   // !FIRST: h-in [t][b][HID] f16 (ws)
    const float*    __restrict__ W_ih,    // (GATES, IN_DIM)
    const float*    __restrict__ W_hh,    // (GATES, HID)
    const float*    __restrict__ b_ih,    // (GATES)
    const float*    __restrict__ b_hh,    // (GATES)
    _Float16*       __restrict__ h_out,   // !LAST: [t][b][HID] f16 (ws)
    const float*    __restrict__ W_fc,    // LAST: (1, HID)
    const float*    __restrict__ b_fc,    // LAST: (1)
    float*          __restrict__ out)     // LAST: (B, 1) f32
{
  __shared__ _Float16 sH[BTILE][HID];        //  4 KB  hidden state (f16)
  __shared__ _Float16 sX[2][BTILE][IN_PAD];  // <=8 KB double-buffered input tile
  __shared__ float    sC[BTILE][HID];        //  8 KB  cell state (f32)
  __shared__ float    sG[GATES * SGPITCH];   // 40 KB  gates [gate][batch] p=20

  const int tid      = threadIdx.x;
  const int wave     = tid >> 5;
  const int lane     = tid & 31;
  const int lane16   = lane & 15;
  const int laneHalf = lane >> 4;
  const int b0       = blockIdx.x * BTILE;

  // ---- Preload weights as register-resident B-fragments (f16), once. ----
  // B-matrix (KxN) mirrors A layout: lane n = lane%16 holds column N (= a row
  // of W, since gates = h @ W^T), halves cover K with the same K pattern.
  float biasr[NTPW];
  v16h  bWhh[NTPW][HID / 32];
  v16h  bWih[NTPW][IN_PAD / 32];

#pragma unroll
  for (int nt = 0; nt < NTPW; ++nt) {
    const int n = wave * (NTPW * 16) + nt * 16 + lane16;  // global gate index
    biasr[nt] = b_ih[n] + b_hh[n];
#pragma unroll
    for (int kc = 0; kc < HID / 32; ++kc) {
      v16h f;
#pragma unroll
      for (int hi = 0; hi < 16; ++hi) {
        int k = kc * 32 + (hi < 8 ? hi + laneHalf * 8
                                  : 16 + (hi - 8) + laneHalf * 8);
        f[hi] = (_Float16)W_hh[n * HID + k];
      }
      bWhh[nt][kc] = f;
    }
#pragma unroll
    for (int kc = 0; kc < IN_PAD / 32; ++kc) {
      v16h f;
#pragma unroll
      for (int hi = 0; hi < 16; ++hi) {
        int k = kc * 32 + (hi < 8 ? hi + laneHalf * 8
                                  : 16 + (hi - 8) + laneHalf * 8);
        _Float16 v = (_Float16)0.0f;
        if (k < IN_DIM) v = (_Float16)W_ih[n * IN_DIM + k];
        f[hi] = v;
      }
      bWih[nt][kc] = f;
    }
  }

  // ---- Zero-init h, c, and the K-padding columns of both sX buffers. ----
  for (int idx = tid; idx < BTILE * HID; idx += NTHREADS) {
    (&sH[0][0])[idx] = (_Float16)0.0f;
    (&sC[0][0])[idx] = 0.0f;
  }
  if (IN_PAD > IN_DIM) {
    for (int idx = tid; idx < 2 * BTILE * (IN_PAD - IN_DIM); idx += NTHREADS) {
      int p = idx / (BTILE * (IN_PAD - IN_DIM));
      int r = idx - p * (BTILE * (IN_PAD - IN_DIM));
      int m = r / (IN_PAD - IN_DIM);
      int i = r % (IN_PAD - IN_DIM);
      sX[p][m][IN_DIM + i] = (_Float16)0.0f;
    }
  }

  // ---- Stage x_0 into buffer 0 (one 8B vector per thread for f16 path). ----
  if (FIRST) {
    for (int idx = tid; idx < BTILE * IN_DIM; idx += NTHREADS) {
      int m = idx / IN_DIM, i = idx - m * IN_DIM;
      sX[0][m][i] = (_Float16)x_f32[((size_t)(b0 + m) * TSTEPS + 0) * IN_DIM + i];
    }
  } else {
    int m = tid >> 5, i = (tid & 31) * 4;   // 512 threads x 4 halfs = 16x128
    *(uint2*)&sX[0][m][i] =
        *(const uint2*)&x_f16[((size_t)0 * BATCH + b0 + m) * HID + i];
  }
  __syncthreads();

  // ---- Time recurrence: 2 barriers per step, x staged one step ahead. ----
  for (int t = 0; t < TSTEPS; ++t) {
    const int pb = t & 1;

    // Hoist A-fragments, then run the WMMA chain (alternating accumulators).
    v16h ax[IN_PAD / 32], ah[HID / 32];
#pragma unroll
    for (int kc = 0; kc < IN_PAD / 32; ++kc)
      ax[kc] = load_a_frag(&sX[pb][0][0], IN_PAD, lane16, kc * 32, laneHalf);
#pragma unroll
    for (int kc = 0; kc < HID / 32; ++kc)
      ah[kc] = load_a_frag(&sH[0][0], HID, lane16, kc * 32, laneHalf);

    // gates = bias + x_t @ W_ih^T + h @ W_hh^T  (f32 accumulation on WMMA).
    v8f acc[NTPW];
#pragma unroll
    for (int nt = 0; nt < NTPW; ++nt)
#pragma unroll
      for (int j = 0; j < 8; ++j) acc[nt][j] = biasr[nt];

#pragma unroll
    for (int kc = 0; kc < IN_PAD / 32; ++kc)
#pragma unroll
      for (int nt = 0; nt < NTPW; ++nt)
        acc[nt] = wmma_f16(ax[kc], bWih[nt][kc], acc[nt]);
#pragma unroll
    for (int kc = 0; kc < HID / 32; ++kc)
#pragma unroll
      for (int nt = 0; nt < NTPW; ++nt)
        acc[nt] = wmma_f16(ah[kc], bWhh[nt][kc], acc[nt]);

    // Stage x_{t+1} into the other buffer: these global loads overlap the
    // scatter barrier and the whole activation phase (off the critical path).
    if (t + 1 < TSTEPS) {
      if (FIRST) {
        for (int idx = tid; idx < BTILE * IN_DIM; idx += NTHREADS) {
          int m = idx / IN_DIM, i = idx - m * IN_DIM;
          sX[pb ^ 1][m][i] =
              (_Float16)x_f32[((size_t)(b0 + m) * TSTEPS + (t + 1)) * IN_DIM + i];
        }
      } else {
        int m = tid >> 5, i = (tid & 31) * 4;
        *(uint2*)&sX[pb ^ 1][m][i] =
            *(const uint2*)&x_f16[((size_t)(t + 1) * BATCH + b0 + m) * HID + i];
        if (t + 2 < TSTEPS && tid < BTILE)   // hint t+2 (global_prefetch_b8)
          __builtin_prefetch(&x_f16[((size_t)(t + 2) * BATCH + b0 + tid) * HID],
                             0, 0);
      }
    }

    // Scatter gate pre-activations. sG is [gate][batch-row] (pitch 20 floats):
    // the 8 contiguous C-fragment floats become two ds_store_b128 per N-tile.
#pragma unroll
    for (int nt = 0; nt < NTPW; ++nt) {
      int n = wave * (NTPW * 16) + nt * 16 + lane16;
      union { v8f v; uint4 q[2]; } g;
      g.v = acc[nt];
      float* p = &sG[n * SGPITCH + laneHalf * 8];
      *(uint4*)(p)     = g.q[0];
      *(uint4*)(p + 4) = g.q[1];
    }
    __syncthreads();

    // Element-wise LSTM cell update: 4 contiguous hidden cols per thread.
    {
      int m  = tid >> 5;                 // batch row 0..15
      int jb = (tid & 31) * 4;           // hidden col base
      float4 cv = *(float4*)&sC[m][jb];
      float  cr[4] = {cv.x, cv.y, cv.z, cv.w};
      _Float16 hv[4];
#pragma unroll
      for (int jj = 0; jj < 4; ++jj) {
        int   j  = jb + jj;
        float ig = fast_sigmoid(sG[(0 * HID + j) * SGPITCH + m]);
        float fg = fast_sigmoid(sG[(1 * HID + j) * SGPITCH + m]);
        float gg = fast_tanh   (sG[(2 * HID + j) * SGPITCH + m]);
        float og = fast_sigmoid(sG[(3 * HID + j) * SGPITCH + m]);
        float c  = fg * cr[jj] + ig * gg;
        cr[jj]   = c;
        hv[jj]   = (_Float16)(og * fast_tanh(c));
      }
      *(float4*)&sC[m][jb] = make_float4(cr[0], cr[1], cr[2], cr[3]);
      *(uint2*)&sH[m][jb]  = *(uint2*)hv;                // 8B packed f16 store
      if (!LAST)
        *(uint2*)&h_out[((size_t)t * BATCH + b0 + m) * HID + jb] =
            *(uint2*)hv;                                 // coalesced 8B store
    }
    __syncthreads();
  }

  // ---- Fused final FC (last layer only, on the final h). ----
  if (LAST) {
    if (tid < BTILE) {
      float a = b_fc[0];
#pragma unroll 8
      for (int j = 0; j < HID; ++j) a += (float)sH[tid][j] * W_fc[j];
      out[b0 + tid] = a;
    }
  }
}

extern "C" void kernel_launch(void* const* d_in, const int* in_sizes, int n_in,
                              void* d_out, int out_size, void* d_ws, size_t ws_size,
                              hipStream_t stream) {
  const float* x     = (const float*)d_in[0];
  const float* W_ih0 = (const float*)d_in[1];
  const float* W_hh0 = (const float*)d_in[2];
  const float* b_ih0 = (const float*)d_in[3];
  const float* b_hh0 = (const float*)d_in[4];
  const float* W_ih1 = (const float*)d_in[5];
  const float* W_hh1 = (const float*)d_in[6];
  const float* b_ih1 = (const float*)d_in[7];
  const float* b_hh1 = (const float*)d_in[8];
  const float* W_fc  = (const float*)d_in[9];
  const float* b_fc  = (const float*)d_in[10];
  float* out = (float*)d_out;

  // Inter-layer hidden states h1: [t][b][HID] f16 = 128*2048*128*2 = 64 MiB.
  _Float16* h1 = (_Float16*)d_ws;

  dim3 grid(BATCH / BTILE);   // 128 workgroups, one 16-row batch tile each
  dim3 block(NTHREADS);       // 16 wave32s

  lstm_layer_kernel<24, 32, 1, 0><<<grid, block, 0, stream>>>(
      x, nullptr, W_ih0, W_hh0, b_ih0, b_hh0, h1, nullptr, nullptr, nullptr);
  lstm_layer_kernel<128, 128, 0, 1><<<grid, block, 0, stream>>>(
      nullptr, h1, W_ih1, W_hh1, b_ih1, b_hh1, nullptr, W_fc, b_fc, out);
}
